// LocalWindowAttention_49959059587809
// MI455X (gfx1250) — matrix operand order repaired and checked by reading, not compiled, not run on hardware
//
#include <hip/hip_runtime.h>
#include <hip/hip_bf16.h>

typedef __attribute__((ext_vector_type(16))) __bf16 v16bf;
typedef __attribute__((ext_vector_type(8)))  __bf16 v8bf;
typedef __attribute__((ext_vector_type(8)))  float  v8f;

#define EPSV 1e-5f
#define SCALEV 0.125f   // 64^-0.5

__device__ __forceinline__ v8f wmma_bf16(v16bf a, v16bf b, v8f c) {
  return __builtin_amdgcn_wmma_f32_16x16x32_bf16(false, a, false, b, (short)0, c,
                                                 false, false);
}

// ---- WMMA fragment loads (ISA 7.12.2 layouts, wave32) ----
// A 16x32 bf16, source stored row-major [m][k] (k contiguous): 2x 16B loads
__device__ __forceinline__ v16bf frag_a_row(const __bf16* P, int ld, int m0, int k0) {
  int lane = threadIdx.x & 31;
  const __bf16* p = P + (m0 + (lane & 15)) * ld + k0 + (((lane >> 4) & 1) << 3);
  v8bf lo = *(const v8bf*)p;
  v8bf hi = *(const v8bf*)(p + 16);
  v16bf a;
#pragma unroll
  for (int e = 0; e < 8; ++e) { a[e] = lo[e]; a[e + 8] = hi[e]; }
  return a;
}

// Same A fragment from row-major fp32 global weights (vectorized float4 loads)
__device__ __forceinline__ v16bf frag_a_f32g(const float* W, int ld, int m0, int k0) {
  int lane = threadIdx.x & 31;
  const float* p = W + (m0 + (lane & 15)) * ld + k0 + (((lane >> 4) & 1) << 3);
  float4 f0 = *(const float4*)(p);
  float4 f1 = *(const float4*)(p + 4);
  float4 f2 = *(const float4*)(p + 16);
  float4 f3 = *(const float4*)(p + 20);
  v16bf a;
  a[0] = (__bf16)f0.x; a[1]  = (__bf16)f0.y; a[2]  = (__bf16)f0.z; a[3]  = (__bf16)f0.w;
  a[4] = (__bf16)f1.x; a[5]  = (__bf16)f1.y; a[6]  = (__bf16)f1.z; a[7]  = (__bf16)f1.w;
  a[8] = (__bf16)f2.x; a[9]  = (__bf16)f2.y; a[10] = (__bf16)f2.z; a[11] = (__bf16)f2.w;
  a[12]= (__bf16)f3.x; a[13] = (__bf16)f3.y; a[14] = (__bf16)f3.z; a[15] = (__bf16)f3.w;
  return a;
}

// B 32x16 bf16, source stored TRANSPOSED [n][k] (k contiguous): 2x 16B loads
__device__ __forceinline__ v16bf frag_b_rowT(const __bf16* P, int ld, int n0, int k0) {
  int lane = threadIdx.x & 31;
  const __bf16* p = P + (n0 + (lane & 15)) * ld + k0 + (((lane >> 4) & 1) << 3);
  v8bf lo = *(const v8bf*)p;
  v8bf hi = *(const v8bf*)(p + 16);
  v16bf b;
#pragma unroll
  for (int e = 0; e < 8; ++e) { b[e] = lo[e]; b[e + 8] = hi[e]; }
  return b;
}

// One block = one 7x7 window; 256 threads = 8 wave32s.
__global__ __launch_bounds__(256)
void lwa_window_kernel(const float* __restrict__ x,
                       const float* __restrict__ qkv_w,
                       const float* __restrict__ qkv_g, const float* __restrict__ qkv_b,
                       const float* __restrict__ qkv_m, const float* __restrict__ qkv_v,
                       const float* __restrict__ dw_w,
                       const float* __restrict__ dw_g, const float* __restrict__ dw_b,
                       const float* __restrict__ dw_m, const float* __restrict__ dw_v,
                       const float* __restrict__ proj_w,
                       const float* __restrict__ proj_g, const float* __restrict__ proj_b,
                       const float* __restrict__ proj_m, const float* __restrict__ proj_v,
                       float* __restrict__ out) {
  // LDS tensors stored in the order their WMMA consumer wants:
  __shared__ __align__(32) __bf16 s_featT[64 * 64];  // feat^T [n][c]  (B of qkv)
  __shared__ __align__(32) __bf16 s_qp[64 * 121];    // q, zero-padded 11x11 halo
  __shared__ __align__(32) __bf16 s_kT[64 * 64];     // k^T [m][c]     (B of scores)
  __shared__ __align__(32) __bf16 s_v[64 * 64];      // v [c][m]       (A of out)
  __shared__ __align__(32) __bf16 s_q2T[64 * 64];    // q2^T [n][c]; aliased attn
  __shared__ __align__(32) float  s_scr[64 * 64];    // S^T f32 / dw wts / BN folds
  __shared__ __align__(32) __bf16 s_catT[64 * 256];  // relu(out)^T [n][cfull]

  __bf16* s_attn = s_q2T;              // attn [n][m] (B of out GEMM)

  const int tid  = threadIdx.x;
  const int wave = tid >> 5;
  const int lane = tid & 31;
  const int nl   = lane & 15;              // D-tile: N index
  const int mh   = ((lane >> 4) & 1) << 3; // D-tile: M base (0 or 8)

  const int wid = blockIdx.x;          // 2048 windows
  const int bat = wid >> 6;
  const int wy  = (wid >> 3) & 7;
  const int wx  = wid & 7;
  const long xbase = (long)bat * 256 * 56 * 56 + (long)(wy * 7) * 56 + wx * 7;

  // ---- Stage A: chunk0 -> s_featT; zero the conv halo buffer once ----
  for (int idx = tid; idx < 64 * 121; idx += 256) s_qp[idx] = (__bf16)0.0f;
  for (int idx = tid; idx < 64 * 64; idx += 256) {
    int c = idx >> 6, n = idx & 63;
    float v = 0.0f;
    if (n < 49) v = x[xbase + (long)c * 3136 + (n / 7) * 56 + (n % 7)];
    s_featT[n * 64 + c] = (__bf16)v;
  }
  __syncthreads();

  for (int head = 0; head < 4; ++head) {
    const float* Wqkv = qkv_w + head * 192 * 64;
    const int ph = head * 192;
    const int dh = head * 64;

    // ---- Stage B0: fold qkv BN into scale/bias (scores buffer is idle) ----
    for (int o = tid; o < 192; o += 256) {
      float inv = qkv_g[ph + o] * rsqrtf(qkv_v[ph + o] + EPSV);
      s_scr[o] = inv;
      s_scr[192 + o] = qkv_b[ph + o] - qkv_m[ph + o] * inv;
    }
    __syncthreads();

    // ---- Stage B: y = BN(Wqkv * feat), 192x64, K=64 ----
    for (int t = wave; t < 48; t += 8) {
      int m0 = (t >> 2) << 4, n0 = (t & 3) << 4;
      v8f acc = {};
#pragma unroll
      for (int kc = 0; kc < 64; kc += 32) {
        v16bf a = frag_a_f32g(Wqkv, 64, m0, kc);
        v16bf b = frag_b_rowT(s_featT, 64, n0, kc);
        acc = wmma_bf16(a, b, acc);
      }
      int n = n0 + nl;
      if (m0 < 64) {                       // q: scatter into padded halo interior
        if (n < 49) {
          int qoff = (n / 7) * 11 + (n % 7) + 24;   // (r+2)*11 + (cc+2)
#pragma unroll
          for (int r = 0; r < 8; ++r) {
            int o = m0 + mh + r;
            s_qp[o * 121 + qoff] = (__bf16)(acc[r] * s_scr[o] + s_scr[192 + o]);
          }
        }
      } else if (m0 < 128) {               // k^T: 8 rows contiguous -> 16B store
        v8bf pk;
#pragma unroll
        for (int r = 0; r < 8; ++r) {
          int o = m0 + mh + r;
          pk[r] = (__bf16)(acc[r] * s_scr[o] + s_scr[192 + o]);
        }
        *(v8bf*)&s_kT[n * 64 + (m0 - 64) + mh] = pk;
      } else {                             // v: [c][m]
#pragma unroll
        for (int r = 0; r < 8; ++r) {
          int o = m0 + mh + r;
          s_v[(m0 - 128 + mh + r) * 64 + n] = (__bf16)(acc[r] * s_scr[o] + s_scr[192 + o]);
        }
      }
    }
    __syncthreads();

    // ---- Stage C: stage dw weights + folded BN, branch-free 5x5 conv ----
    for (int idx = tid; idx < 64 * 25; idx += 256)
      s_scr[idx] = dw_w[head * 64 * 25 + idx];
    if (tid < 64) {
      float inv = dw_g[dh + tid] * rsqrtf(dw_v[dh + tid] + EPSV);
      s_scr[1600 + tid] = inv;
      s_scr[1664 + tid] = dw_b[dh + tid] - dw_m[dh + tid] * inv;
    }
    __syncthreads();
    for (int idx = tid; idx < 64 * 64; idx += 256) {
      int c = idx >> 6, n = idx & 63;
      float val = 0.0f;
      if (n < 49) {
        const __bf16* qp = s_qp + c * 121 + (n / 7) * 11 + (n % 7); // 5x5 top-left
        const float*  wp = s_scr + c * 25;
        float a = 0.0f;
#pragma unroll
        for (int dr = 0; dr < 5; ++dr)
#pragma unroll
          for (int dc = 0; dc < 5; ++dc)
            a += wp[dr * 5 + dc] * (float)qp[dr * 11 + dc];
        val = a * s_scr[1600 + c] + s_scr[1664 + c];
      }
      s_q2T[n * 64 + c] = (__bf16)val;
    }
    __syncthreads();

    // ---- Stage D: S^T[m][n] = scale * (q2^T k)^T, packed f32 stores ----
    for (int t = wave; t < 16; t += 8) {
      int m0 = (t >> 2) << 4, n0 = (t & 3) << 4;   // m0: S-row block, n0: S-col block
      v8f acc = {};
#pragma unroll
      for (int kc = 0; kc < 64; kc += 32) {
        v16bf a = frag_a_row(s_q2T, 64, m0, kc);  // A(n,c) row-major [n][c]
        v16bf b = frag_b_rowT(s_kT, 64, n0, kc);  // B(c,m) via k^T [m][c]
        acc = wmma_bf16(a, b, acc);
      }
      float* dst = &s_scr[(n0 + nl) * 64 + m0 + mh];
      float4 lo = make_float4(acc[0] * SCALEV, acc[1] * SCALEV,
                              acc[2] * SCALEV, acc[3] * SCALEV);
      float4 hi = make_float4(acc[4] * SCALEV, acc[5] * SCALEV,
                              acc[6] * SCALEV, acc[7] * SCALEV);
      *(float4*)dst = lo;
      *(float4*)(dst + 4) = hi;
    }
    __syncthreads();

    // ---- Stage E: softmax over m<49 (S^T: bank-conflict-free) -> attn [n][m] ----
    if (tid < 64) {
      int row = tid;
      if (row < 49) {
        float mx = -1e30f;
        for (int m = 0; m < 49; ++m) mx = fmaxf(mx, s_scr[m * 64 + row]);
        float sum = 0.0f;
        for (int m = 0; m < 49; ++m) sum += __expf(s_scr[m * 64 + row] - mx);
        float inv = 1.0f / sum;
        for (int m = 0; m < 64; ++m) {
          float p = (m < 49) ? __expf(s_scr[m * 64 + row] - mx) * inv : 0.0f;
          s_attn[row * 64 + m] = (__bf16)p;
        }
      } else {
        for (int m = 0; m < 64; ++m) s_attn[row * 64 + m] = (__bf16)0.0f;
      }
    }
    __syncthreads();

    // ---- Stage F: out[c][n] = v * attn^T; packed catT + featT stores ----
    for (int t = wave; t < 16; t += 8) {
      int m0 = (t >> 2) << 4, n0 = (t & 3) << 4;
      v8f acc = {};
#pragma unroll
      for (int kc = 0; kc < 64; kc += 32) {
        v16bf a = frag_a_row(s_v, 64, m0, kc);     // A(c,m) row-major [c][m]
        v16bf b = frag_b_rowT(s_attn, 64, n0, kc); // B(m,n) via attn [n][m]
        acc = wmma_bf16(a, b, acc);
      }
      int n = n0 + nl;
      v8bf cat_pk;
#pragma unroll
      for (int r = 0; r < 8; ++r) cat_pk[r] = (__bf16)fmaxf(acc[r], 0.0f);
      *(v8bf*)&s_catT[n * 256 + head * 64 + m0 + mh] = cat_pk;
      if (head < 3) {
        v8bf ft_pk;
#pragma unroll
        for (int r = 0; r < 8; ++r) {
          float nxt = acc[r];
          if (n < 49)
            nxt += x[xbase + (long)((head + 1) * 64 + m0 + mh + r) * 3136 +
                     (n / 7) * 56 + (n % 7)];
          ft_pk[r] = (__bf16)nxt;
        }
        *(v8bf*)&s_featT[n * 64 + m0 + mh] = ft_pk;
      }
    }
    __syncthreads();
  }

  // ---- Stage G0: fold proj BN into scale/bias ----
  for (int o = tid; o < 256; o += 256) {
    float inv = proj_g[o] * rsqrtf(proj_v[o] + EPSV);
    s_scr[o] = inv;
    s_scr[256 + o] = proj_b[o] - proj_m[o] * inv;
  }
  __syncthreads();

  // ---- Stage G: proj = BN(Wproj * relu(cat)), 256x64, K=256 ----
  for (int t = wave; t < 64; t += 8) {
    int m0 = (t >> 2) << 4, n0 = (t & 3) << 4;
    v8f acc = {};
#pragma unroll 2
    for (int kc = 0; kc < 256; kc += 32) {
      v16bf a = frag_a_f32g(proj_w, 256, m0, kc);
      v16bf b = frag_b_rowT(s_catT, 256, n0, kc);
      acc = wmma_bf16(a, b, acc);
    }
    int n = n0 + nl;
    if (n < 49) {
      long obase = xbase + (n / 7) * 56 + (n % 7);
#pragma unroll
      for (int r = 0; r < 8; ++r) {
        int o = m0 + mh + r;
        out[obase + (long)o * 3136] = acc[r] * s_scr[o] + s_scr[256 + o];
      }
    }
  }
}

extern "C" void kernel_launch(void* const* d_in, const int* in_sizes, int n_in,
                              void* d_out, int out_size, void* d_ws, size_t ws_size,
                              hipStream_t stream) {
  (void)in_sizes; (void)n_in; (void)out_size; (void)d_ws; (void)ws_size;
  const float* x      = (const float*)d_in[0];
  const float* qkv_w  = (const float*)d_in[1];
  const float* qkv_g  = (const float*)d_in[2];
  const float* qkv_b  = (const float*)d_in[3];
  const float* qkv_m  = (const float*)d_in[4];
  const float* qkv_v  = (const float*)d_in[5];
  const float* dw_w   = (const float*)d_in[6];
  const float* dw_g   = (const float*)d_in[7];
  const float* dw_b   = (const float*)d_in[8];
  const float* dw_m   = (const float*)d_in[9];
  const float* dw_v   = (const float*)d_in[10];
  const float* proj_w = (const float*)d_in[11];
  const float* proj_g = (const float*)d_in[12];
  const float* proj_b = (const float*)d_in[13];
  const float* proj_m = (const float*)d_in[14];
  const float* proj_v = (const float*)d_in[15];
  float* out = (float*)d_out;

  dim3 grid(2048), block(256);
  hipLaunchKernelGGL(lwa_window_kernel, grid, block, 0, stream,
                     x, qkv_w, qkv_g, qkv_b, qkv_m, qkv_v,
                     dw_w, dw_g, dw_b, dw_m, dw_v,
                     proj_w, proj_g, proj_b, proj_m, proj_v, out);
}
